// InteractionNetwork_attention_65085934403606
// MI455X (gfx1250) — matrix operand order repaired and checked by reading, compiled, not run on hardware
//
#include <hip/hip_runtime.h>

// ---------------------------------------------------------------------------
// MI455X (gfx1250, wave32) graph interaction network.
// GEMMs on the matrix core (v_wmma_f32_16x16x32_bf16, f32 accumulate).
// Weights DMA'd into LDS by the Tensor Data Mover (tensor_load_to_lds,
// TENSORcnt) and shared by the block's 8 waves; QKV fused so the activation
// tile is read once for 3 weights; LayerNorm+residual fused into the final
// MLP GEMM epilogue; edge attention fused with the node attention edge-score
// pass. Softmax/segment ops are fp32 VALU + L2 atomics.
// ---------------------------------------------------------------------------

typedef __bf16 bf16_t;
typedef float    v8f   __attribute__((ext_vector_type(8)));
typedef bf16_t   v16bf __attribute__((ext_vector_type(16)));
typedef unsigned u32x4 __attribute__((ext_vector_type(4)));
typedef int      i32x4 __attribute__((ext_vector_type(4)));
typedef int      i32x8 __attribute__((ext_vector_type(8)));

#define HD  8     // heads
#define DIM 128   // feature dim
#define DH  16    // per-head dim

#if __has_builtin(__builtin_amdgcn_tensor_load_to_lds) && \
    __has_builtin(__builtin_amdgcn_s_wait_tensorcnt)
#define GINET_TDM 1
#endif

// ---------------------------------------------------------------------------
// Weight pre-pack: fp32 [K=128][N=128] row-major -> bf16 fragments in the
// V_WMMA 16-bit B-matrix (32x16) lane layout:
//   lane l holds column n = l&15 ; K-half h = l>>4
//   element i (0..15): K = kc*32 + h*8 + (i&7) + (i>>3)*16
// stored as dst[((nt*4+kc)*32+lane)*16 + i]  (16 contiguous bf16 = 32B/lane)
// ---------------------------------------------------------------------------
struct W12 { const float* w[12]; };

__global__ __launch_bounds__(256)
void ginet_pack_w(W12 srcs, bf16_t* __restrict__ dst) {
  int idx = blockIdx.x * blockDim.x + threadIdx.x;
  if (idx >= 12 * 16384) return;
  int m    = idx >> 14;
  int r    = idx & 16383;
  int i    = r & 15;
  int lane = (r >> 4) & 31;
  int kc   = (r >> 9) & 3;
  int nt   = r >> 11;
  int K    = kc * 32 + (lane >> 4) * 8 + (i & 7) + (i >> 3) * 16;
  int N    = nt * 16 + (lane & 15);
  dst[idx] = (bf16_t)srcs.w[m][K * DIM + N];
}

// ---------------------------------------------------------------------------
// Tensor Data Mover staging of one 32KB packed weight matrix into LDS.
// D# descriptor: group0: count=1, lds_addr=0 (block's single __shared__
// buffer), global_addr at bits[120:64], type=2; group1: data_size=3 (8B),
// tensor_dim0=4096, tensor_dim1=1, tile_dim0=4096 (=32KB), stride=4096.
// Issued by wave 0 only (one DMA per block); wave 0 waits on TENSORcnt; all
// waves then pass a block barrier and execute lds_publish(), which captures
// the LDS buffer in an opaque asm with a memory clobber so the compiler must
// assume the DMA wrote it (the DMA's LDS stores are invisible to alias
// analysis otherwise).  The actual fragment reads keep using the original
// LDS pointer so they lower to ds_load_b128 (not flat).
// ---------------------------------------------------------------------------
__device__ __forceinline__ void tdm_issue(const bf16_t* __restrict__ Bp) {
#ifdef GINET_TDM
  if (threadIdx.x < 32) {
    unsigned long long ga = (unsigned long long)Bp;
    u32x4 g0;
    g0[0] = 1u;                                               // count=1
    g0[1] = 0u;                                               // lds_addr=0
    g0[2] = (unsigned)ga;                                     // addr[31:0]
    g0[3] = (unsigned)((ga >> 32) & 0x01FFFFFFu) | (2u << 30);// addr[56:32],type=2
    i32x8 g1;
    g1[0] = (int)(3u << 16);        // workgroup_mask=0, data_size=3 (8B)
    g1[1] = (int)(0x1000u << 16);   // tensor_dim0[15:0]=4096 at bits[63:48]
    g1[2] = (int)(1u << 16);        // tensor_dim0 hi=0; tensor_dim1=1
    g1[3] = (int)(0x1000u << 16);   // tensor_dim1 hi=0; tile_dim0=4096
    g1[4] = 0;                      // tile_dim1=0, tile_dim2=0 (unused)
    g1[5] = 0x1000;                 // tensor_dim0_stride=4096
    g1[6] = 0;
    g1[7] = 0;
    i32x4 gz4 = {0, 0, 0, 0};
    i32x8 gz8 = {0, 0, 0, 0, 0, 0, 0, 0};
    __builtin_amdgcn_tensor_load_to_lds(g0, g1, gz4, gz4, gz8, 0);
  }
#else
  (void)Bp;
#endif
}

__device__ __forceinline__ void tdm_wait() {
#ifdef GINET_TDM
  if (threadIdx.x < 32) __builtin_amdgcn_s_wait_tensorcnt(0);
#endif
}

__device__ __forceinline__ void stage_copy(const bf16_t* __restrict__ Bp,
                                           bf16_t* __restrict__ Bs) {
  const int4* s = (const int4*)Bp;   // 2048 x 16B = 32KB
  int4*       d = (int4*)Bs;
  for (int i = threadIdx.x; i < 2048; i += 256) d[i] = s[i];
}

// Capture the LDS buffer in an opaque asm with a memory clobber: the buffer
// escapes, so the compiler must assume the asm (standing in for the TDM DMA)
// wrote through it.  The pointer itself is NOT used for the loads, so the
// subsequent reads still lower to ds_load (LDS address space preserved).
__device__ __forceinline__ void lds_publish(bf16_t* p) {
  asm volatile("" : "+v"(p) : : "memory");
}

// ---------------------------------------------------------------------------
// WMMA GEMM strip helpers: block = 256 thr = 8 waves; each wave owns a
// 16-row x 128-col strip. A is fp32->bf16 converted on the fly into the ISA
// 16-bit A-matrix 16x32 layout (lane half h: elems 0..7 -> K=kc*32+h*8+i,
// elems 8..15 -> +16).
// ---------------------------------------------------------------------------
__device__ __forceinline__ void load_a_frags(const float* __restrict__ A,
                                             int rA, int h, v16bf afr[4]) {
#pragma unroll
  for (int kc = 0; kc < 4; ++kc) {
    const float* ap = A + (size_t)rA * DIM + kc * 32 + h * 8;
#pragma unroll
    for (int i = 0; i < 8; ++i) afr[kc][i] = (bf16_t)ap[i];
#pragma unroll
    for (int i = 0; i < 8; ++i) afr[kc][8 + i] = (bf16_t)ap[16 + i];
  }
}

__device__ __forceinline__ void mma_strip(const v16bf afr[4],
                                          const bf16_t* Bs,
                                          int lane, v8f acc[8]) {
  const v16bf* Bv = reinterpret_cast<const v16bf*>(Bs);
#pragma unroll
  for (int kc = 0; kc < 4; ++kc) {
#pragma unroll
    for (int nt = 0; nt < 8; ++nt) {
      v16bf b = Bv[(nt * 4 + kc) * 32 + lane];
      acc[nt] = __builtin_amdgcn_wmma_f32_16x16x32_bf16(
          false, afr[kc], false, b, (short)0, acc[nt], false, false);
    }
  }
}

// ---------------------------------------------------------------------------
// Fused QKV: 3 x (C[w] = A @ B[w]).  A-tile fragments loaded/converted ONCE,
// the three weight matrices swapped through one 32KB LDS buffer.
// ---------------------------------------------------------------------------
struct P3 { float* c[3]; };

__global__ __launch_bounds__(256)
void ginet_qkv3(const float* __restrict__ A, const bf16_t* __restrict__ Bp,
                P3 outs, int M) {
  __shared__ __align__(16) bf16_t Bs[16384];
  const int lane = threadIdx.x & 31;
  const int wave = threadIdx.x >> 5;
  const int m0   = blockIdx.x * 128 + wave * 16;
  const int h    = lane >> 4;
  int rA = m0 + (lane & 15);
  if (rA >= M) rA = M - 1;          // clamp: EXEC stays all-ones for WMMA

  v16bf afr[4];
  load_a_frags(A, rA, h, afr);

  const int col0 = lane & 15;
  const int rb   = m0 + h * 8;

  for (int w = 0; w < 3; ++w) {
    __syncthreads();                      // previous readers done with LDS
#ifdef GINET_TDM
    tdm_issue(Bp + w * 16384);
    tdm_wait();
#else
    stage_copy(Bp + w * 16384, Bs);
#endif
    __syncthreads();
    lds_publish(Bs);                      // DMA writes now visible per stage

    v8f acc[8];
#pragma unroll
    for (int t = 0; t < 8; ++t) acc[t] = (v8f){0.f,0.f,0.f,0.f,0.f,0.f,0.f,0.f};
    mma_strip(afr, Bs, lane, acc);

    float* C = outs.c[w];
    if (m0 + 16 <= M) {                   // uniform fast path (M % 16 == 0)
#pragma unroll
      for (int nt = 0; nt < 8; ++nt) {
        int col = nt * 16 + col0;
#pragma unroll
        for (int j = 0; j < 8; ++j)
          C[(size_t)(rb + j) * DIM + col] = acc[nt][j];
      }
    } else {
#pragma unroll
      for (int nt = 0; nt < 8; ++nt) {
        int col = nt * 16 + col0;
#pragma unroll
        for (int j = 0; j < 8; ++j) {
          int row = rb + j;
          if (row < M) C[(size_t)row * DIM + col] = acc[nt][j];
        }
      }
    }
  }
}

// ---------------------------------------------------------------------------
// Single-weight GEMM with epilogue modes:
//   0: C = A@B + bias            1: C = relu(A@B + bias)
//   2: C = LayerNorm(A@B + bias)*g + b + res   (biased var, eps 1e-5)
// Mode 2: a wave holds complete 128-wide rows -> row stats via shfl_xor
// over the 16-lane half (offsets 1,2,4,8).
// ---------------------------------------------------------------------------
__global__ __launch_bounds__(256)
void ginet_gemm128(const float* __restrict__ A, const bf16_t* __restrict__ Bp,
                   const float* __restrict__ bias, float* __restrict__ C,
                   int M, int mode, const float* __restrict__ g,
                   const float* __restrict__ b, const float* __restrict__ res) {
  __shared__ __align__(16) bf16_t Bs[16384];
  const int lane = threadIdx.x & 31;
  const int wave = threadIdx.x >> 5;
  const int m0   = blockIdx.x * 128 + wave * 16;
  const int h    = lane >> 4;
  int rA = m0 + (lane & 15);
  if (rA >= M) rA = M - 1;

  v16bf afr[4];
#ifdef GINET_TDM
  tdm_issue(Bp);                 // DMA overlaps the A-fragment loads below
  load_a_frags(A, rA, h, afr);
  tdm_wait();
#else
  stage_copy(Bp, Bs);
  load_a_frags(A, rA, h, afr);
#endif
  __syncthreads();
  lds_publish(Bs);

  v8f acc[8];
#pragma unroll
  for (int t = 0; t < 8; ++t) acc[t] = (v8f){0.f,0.f,0.f,0.f,0.f,0.f,0.f,0.f};
  mma_strip(afr, Bs, lane, acc);

  const int col0 = lane & 15;
  const int rb   = m0 + h * 8;
  const bool full = (m0 + 16 <= M);      // uniform per wave (M % 16 == 0)
  float bv[8];
#pragma unroll
  for (int nt = 0; nt < 8; ++nt) bv[nt] = bias ? bias[nt * 16 + col0] : 0.f;

  if (mode == 2) {
    // fused LayerNorm + affine + residual
    float sum[8], sq[8];
#pragma unroll
    for (int j = 0; j < 8; ++j) {
      float s = 0.f, q = 0.f;
#pragma unroll
      for (int nt = 0; nt < 8; ++nt) {
        float v = acc[nt][j] + bv[nt];
        s += v; q += v * v;
      }
      sum[j] = s; sq[j] = q;
    }
#pragma unroll
    for (int ofs = 1; ofs < 16; ofs <<= 1) {
#pragma unroll
      for (int j = 0; j < 8; ++j) {
        sum[j] += __shfl_xor(sum[j], ofs, 32);
        sq[j]  += __shfl_xor(sq[j],  ofs, 32);
      }
    }
    if (full) {
#pragma unroll
      for (int j = 0; j < 8; ++j) {
        float mu   = sum[j] * (1.f / DIM);
        float var  = sq[j] * (1.f / DIM) - mu * mu;
        float rstd = rsqrtf(var + 1e-5f);
        size_t ro  = (size_t)(rb + j) * DIM;
#pragma unroll
        for (int nt = 0; nt < 8; ++nt) {
          int col = nt * 16 + col0;
          float v = acc[nt][j] + bv[nt];
          C[ro + col] = (v - mu) * rstd * g[col] + b[col] + res[ro + col];
        }
      }
    } else {
#pragma unroll
      for (int j = 0; j < 8; ++j) {
        int row = rb + j;
        if (row < M) {
          float mu   = sum[j] * (1.f / DIM);
          float var  = sq[j] * (1.f / DIM) - mu * mu;
          float rstd = rsqrtf(var + 1e-5f);
#pragma unroll
          for (int nt = 0; nt < 8; ++nt) {
            int col = nt * 16 + col0;
            float v = acc[nt][j] + bv[nt];
            C[(size_t)row * DIM + col] =
                (v - mu) * rstd * g[col] + b[col] + res[(size_t)row * DIM + col];
          }
        }
      }
    }
  } else {
    if (full) {
#pragma unroll
      for (int nt = 0; nt < 8; ++nt) {
        int col = nt * 16 + col0;
#pragma unroll
        for (int j = 0; j < 8; ++j) {
          float v = acc[nt][j] + bv[nt];
          if (mode == 1) v = fmaxf(v, 0.f);
          C[(size_t)(rb + j) * DIM + col] = v;
        }
      }
    } else {
#pragma unroll
      for (int nt = 0; nt < 8; ++nt) {
        int col = nt * 16 + col0;
#pragma unroll
        for (int j = 0; j < 8; ++j) {
          int row = rb + j;
          if (row < M) {
            float v = acc[nt][j] + bv[nt];
            if (mode == 1) v = fmaxf(v, 0.f);
            C[(size_t)row * DIM + col] = v;
          }
        }
      }
    }
  }
}

// ---------------------------------------------------------------------------
// monotonic float <-> uint encoding for atomicMax-based segment max
// ---------------------------------------------------------------------------
__device__ __forceinline__ unsigned f2ord(float f) {
  unsigned u = __float_as_uint(f);
  return (u & 0x80000000u) ? ~u : (u | 0x80000000u);
}
__device__ __forceinline__ float ord2f(unsigned u) {
  return __uint_as_float((u & 0x80000000u) ? (u ^ 0x80000000u) : ~u);
}
__device__ __forceinline__ void atomicAddF(float* p, float v) {
  __hip_atomic_fetch_add(p, v, __ATOMIC_RELAXED, __HIP_MEMORY_SCOPE_AGENT);
}

// ---------------------------------------------------------------------------
// Fused per-(edge,head) pass:
//  (a) edge update: 3-way softmax over {edge-self, recv-node, send-node},
//      writes agg_e
//  (b) node attention edge score b2 = <xq[recv], ek>*scale, segment max via
//      ordered-uint atomicMax (mx must be pre-initialized by ginet_node_b1)
// Shares the ek / edge_index loads between the two computations.
// ---------------------------------------------------------------------------
__global__ __launch_bounds__(256)
void ginet_edge_fused(const float* __restrict__ eq, const float* __restrict__ ek,
                      const float* __restrict__ ev, const float* __restrict__ xq,
                      const float* __restrict__ xk, const float* __restrict__ xv,
                      const int* __restrict__ ei, float* __restrict__ agg,
                      float* __restrict__ b2, unsigned* __restrict__ mx, int E) {
  int idx = blockIdx.x * blockDim.x + threadIdx.x;
  if (idx >= E * HD) return;
  int e = idx >> 3, hh = idx & 7;
  int send = ei[e], recv = ei[E + e];
  size_t eo = (size_t)e * DIM + hh * DH;
  size_t ro = (size_t)recv * DIM + hh * DH;
  size_t so = (size_t)send * DIM + hh * DH;
  float d0 = 0.f, d1 = 0.f, d2 = 0.f, db = 0.f;
#pragma unroll
  for (int j = 0; j < DH; ++j) {
    float q  = eq[eo + j];
    float kk = ek[eo + j];
    d0 += q * kk;
    d1 += q * xk[ro + j];
    d2 += q * xk[so + j];
    db += xq[ro + j] * kk;
  }
  const float s = 0.25f;  // 1/sqrt(16)
  d0 *= s; d1 *= s; d2 *= s; db *= s;
  float m  = fmaxf(d0, fmaxf(d1, d2));
  float w0 = __expf(d0 - m), w1 = __expf(d1 - m), w2 = __expf(d2 - m);
  float inv = 1.f / (w0 + w1 + w2);
  w0 *= inv; w1 *= inv; w2 *= inv;
#pragma unroll
  for (int j = 0; j < DH; ++j)
    agg[eo + j] = w0 * ev[eo + j] + w1 * xv[ro + j] + w2 * xv[so + j];
  b2[idx] = db;
  atomicMax(&mx[recv * HD + hh], f2ord(db));
}

// ---- node segment softmax pipeline ----------------------------------------
__global__ __launch_bounds__(256)
void ginet_node_b1(const float* __restrict__ xq, const float* __restrict__ xk,
                   float* __restrict__ b1, unsigned* __restrict__ mx, int N) {
  int idx = blockIdx.x * blockDim.x + threadIdx.x;
  if (idx >= N * HD) return;
  int n = idx >> 3, hh = idx & 7;
  size_t o = (size_t)n * DIM + hh * DH;
  float d = 0.f;
#pragma unroll
  for (int j = 0; j < DH; ++j) d += xq[o + j] * xk[o + j];
  d *= 0.25f;
  b1[idx] = d;
  mx[idx] = f2ord(d);   // init segment-max with node-self score
}

__global__ __launch_bounds__(256)
void ginet_node_e1(float* __restrict__ b1, const unsigned* __restrict__ mx,
                   float* __restrict__ denom, int N) {
  int idx = blockIdx.x * blockDim.x + threadIdx.x;
  if (idx >= N * HD) return;
  float e1 = __expf(b1[idx] - ord2f(mx[idx]));
  b1[idx]  = e1;
  denom[idx] = e1;
}

__global__ __launch_bounds__(256)
void ginet_node_e2(float* __restrict__ b2, const unsigned* __restrict__ mx,
                   const int* __restrict__ ei, float* __restrict__ denom, int E) {
  int idx = blockIdx.x * blockDim.x + threadIdx.x;
  if (idx >= E * HD) return;
  int e = idx >> 3, hh = idx & 7;
  int recv = ei[E + e];
  float e2 = __expf(b2[idx] - ord2f(mx[recv * HD + hh]));
  b2[idx]  = e2;
  atomicAddF(&denom[recv * HD + hh], e2);
}

__global__ __launch_bounds__(256)
void ginet_node_agg_init(const float* __restrict__ e1, const float* __restrict__ denom,
                         const float* __restrict__ xv, float* __restrict__ agg, int N) {
  int idx = blockIdx.x * blockDim.x + threadIdx.x;
  if (idx >= N * HD) return;
  int n = idx >> 3, hh = idx & 7;
  float w = e1[idx] / denom[idx];
  size_t o = (size_t)n * DIM + hh * DH;
#pragma unroll
  for (int j = 0; j < DH; ++j) agg[o + j] = w * xv[o + j];
}

__global__ __launch_bounds__(256)
void ginet_node_agg_scatter(const float* __restrict__ e2, const float* __restrict__ denom,
                            const float* __restrict__ ev, const int* __restrict__ ei,
                            float* __restrict__ agg, int E) {
  int idx = blockIdx.x * blockDim.x + threadIdx.x;
  if (idx >= E * HD) return;
  int e = idx >> 3, hh = idx & 7;
  int recv = ei[E + e];
  float w = e2[idx] / denom[recv * HD + hh];
  size_t eo = (size_t)e * DIM + hh * DH;
  size_t ro = (size_t)recv * DIM + hh * DH;
#pragma unroll
  for (int j = 0; j < DH; ++j) atomicAddF(&agg[ro + j], w * ev[eo + j]);
}

// ---------------------------------------------------------------------------
extern "C" void kernel_launch(void* const* d_in, const int* in_sizes, int n_in,
                              void* d_out, int out_size, void* d_ws, size_t ws_size,
                              hipStream_t stream) {
  const float* x    = (const float*)d_in[0];
  const float* ef   = (const float*)d_in[1];
  const float* wq_n = (const float*)d_in[2];
  const float* wk_n = (const float*)d_in[3];
  const float* wv_n = (const float*)d_in[4];
  const float* wq_e = (const float*)d_in[5];
  const float* wk_e = (const float*)d_in[6];
  const float* wv_e = (const float*)d_in[7];
  const float* nmw  = (const float*)d_in[8];   // (3,128,128)
  const float* nmb  = (const float*)d_in[9];   // (3,128)
  const float* nlg  = (const float*)d_in[10];
  const float* nlb  = (const float*)d_in[11];
  const float* emw  = (const float*)d_in[12];
  const float* emb  = (const float*)d_in[13];
  const float* elg  = (const float*)d_in[14];
  const float* elb  = (const float*)d_in[15];
  const int*   ei   = (const int*)d_in[16];

  const int N = in_sizes[0] / DIM;   // 50000
  const int E = in_sizes[1] / DIM;   // 400000

  // ---- workspace carve ----
  size_t off = 0;
  auto carve = [&](size_t bytes) -> void* {
    void* p = (char*)d_ws + off;
    off += (bytes + 255) & ~(size_t)255;
    return p;
  };
  bf16_t*   wpk  = (bf16_t*)carve(12 * 16384 * sizeof(bf16_t));
  float*    xq   = (float*)carve((size_t)N * DIM * 4);   // reused: node h1
  float*    xk   = (float*)carve((size_t)N * DIM * 4);   // reused: node h2
  float*    xv   = (float*)carve((size_t)N * DIM * 4);
  float*    eq   = (float*)carve((size_t)E * DIM * 4);   // reused: edge h1
  float*    ekb  = (float*)carve((size_t)E * DIM * 4);   // reused: edge h2
  float*    evb  = (float*)carve((size_t)E * DIM * 4);
  float*    agge = (float*)carve((size_t)E * DIM * 4);
  float*    aggn = (float*)carve((size_t)N * DIM * 4);
  float*    b1   = (float*)carve((size_t)N * HD * 4);    // -> e1
  float*    b2   = (float*)carve((size_t)E * HD * 4);    // -> e2
  unsigned* mx   = (unsigned*)carve((size_t)N * HD * 4);
  float*    den  = (float*)carve((size_t)N * HD * 4);

  float* out_node = (float*)d_out;
  float* out_edge = out_node + (size_t)N * DIM;

  // ---- 1. pack all 12 weight matrices to WMMA-fragment bf16 ----
  W12 srcs;
  srcs.w[0] = wq_n; srcs.w[1] = wk_n; srcs.w[2] = wv_n;
  srcs.w[3] = wq_e; srcs.w[4] = wk_e; srcs.w[5] = wv_e;
  srcs.w[6] = nmw;  srcs.w[7] = nmw + 16384;  srcs.w[8] = nmw + 32768;
  srcs.w[9] = emw;  srcs.w[10] = emw + 16384; srcs.w[11] = emw + 32768;
  ginet_pack_w<<<(12 * 16384 + 255) / 256, 256, 0, stream>>>(srcs, wpk);

  const int gN = (N + 127) / 128;
  const int gE = (E + 127) / 128;
  const int tN = (N * HD + 255) / 256;
  const int tE = (E * HD + 255) / 256;

  // ---- 2. fused QKV projections (A read once per tensor) ----
  P3 pn; pn.c[0] = xq; pn.c[1] = xk;  pn.c[2] = xv;
  P3 pe; pe.c[0] = eq; pe.c[1] = ekb; pe.c[2] = evb;
  ginet_qkv3<<<gN, 256, 0, stream>>>(x,  wpk,             pn, N);
  ginet_qkv3<<<gE, 256, 0, stream>>>(ef, wpk + 3 * 16384, pe, E);

  // ---- 3. node self-scores + segment-max init (must precede edge_fused) ----
  ginet_node_b1<<<tN, 256, 0, stream>>>(xq, xk, b1, mx, N);

  // ---- 4. fused edge update + node edge-scores (agg_e, b2, segment max) ----
  ginet_edge_fused<<<tE, 256, 0, stream>>>(eq, ekb, evb, xq, xk, xv, ei,
                                           agge, b2, mx, E);

  // ---- 5. node segment softmax finish ----
  ginet_node_e1<<<tN, 256, 0, stream>>>(b1, mx, den, N);
  ginet_node_e2<<<tE, 256, 0, stream>>>(b2, mx, ei, den, E);
  ginet_node_agg_init<<<tN, 256, 0, stream>>>(b1, den, xv, aggn, N);
  ginet_node_agg_scatter<<<tE, 256, 0, stream>>>(b2, den, evb, ei, aggn, E);

  // ---- 6. edge MLP; layer 3 fuses LN + affine + residual -> d_out ----
  ginet_gemm128<<<gE, 256, 0, stream>>>(agge, wpk + 9 * 16384,  emb,       eq,  E, 1,
                                        nullptr, nullptr, nullptr);
  ginet_gemm128<<<gE, 256, 0, stream>>>(eq,   wpk + 10 * 16384, emb + 128, ekb, E, 1,
                                        nullptr, nullptr, nullptr);
  ginet_gemm128<<<gE, 256, 0, stream>>>(ekb,  wpk + 11 * 16384, emb + 256, out_edge, E, 2,
                                        elg, elb, ef);

  // ---- 7. node MLP; layer 3 fuses LN + affine + residual -> d_out ----
  ginet_gemm128<<<gN, 256, 0, stream>>>(aggn, wpk + 6 * 16384, nmb,       xq, N, 1,
                                        nullptr, nullptr, nullptr);
  ginet_gemm128<<<gN, 256, 0, stream>>>(xq,   wpk + 7 * 16384, nmb + 128, xk, N, 1,
                                        nullptr, nullptr, nullptr);
  ginet_gemm128<<<gN, 256, 0, stream>>>(xk,   wpk + 8 * 16384, nmb + 256, out_node, N, 2,
                                        nlg, nlb, x);
}
